// ChunkwiseMultiKeyDeltaAttention_68813966017026
// MI455X (gfx1250) — matrix operand order repaired and checked by reading, not compiled, not run on hardware
//
#include <hip/hip_runtime.h>
#include <hip/hip_bf16.h>
#include <cstdint>

// ---------------------------------------------------------------------------
// Problem constants (from the reference)
// ---------------------------------------------------------------------------
#define B_     2
#define T_     2048
#define D_     2048
#define H_     16
#define DK_    128
#define DV_    128
#define R_     4
#define CHUNK_ 64
#define NCHUNK (T_ / CHUNK_)
#define BT     (B_ * T_)
#define CR_    (CHUNK_ * R_)            // 256
#define SCALE_ 0.08838834764831845f     // DK^-0.5

typedef __attribute__((ext_vector_type(16))) _Float16 v16h;
typedef __attribute__((ext_vector_type(8)))  _Float16 v8h;
typedef __attribute__((ext_vector_type(8)))  float    v8f;

static __device__ __forceinline__ float sigmoidf_(float x) { return 1.f / (1.f + __expf(-x)); }
static __device__ __forceinline__ float softplusf_(float x) { return x > 20.f ? x : log1pf(__expf(x)); }

union FragU { v16h v; v8h h[2]; };

// ---------------------------------------------------------------------------
// CDNA5 async global->LDS copy (CDNA5 ISA).
// Each enabled lane copies 16B from its global address to its LDS address.
// Tracked by ASYNCcnt; per-wave completion via s_wait_asynccnt.
// Generic LDS pointers on gfx1250 carry the LDS byte offset in addr[31:0]
// (ISA §10.2 aperture rules), so truncation yields the VDST LDS address.
// ---------------------------------------------------------------------------
static __device__ __forceinline__ uint32_t lds_off_(const void* p) {
  return (uint32_t)(uintptr_t)p;
}
static __device__ __forceinline__ void async_ld_b128(uint32_t ldsoff, const _Float16* g) {
  asm volatile("global_load_async_to_lds_b128 %0, %1, off"
               :: "v"(ldsoff), "v"(g) : "memory");
}
#define WAIT_ASYNC_0() asm volatile("s_wait_asynccnt 0x0" ::: "memory")
#define WAIT_ASYNC_4() asm volatile("s_wait_asynccnt 0x4" ::: "memory")

// ---------------------------------------------------------------------------
// WMMA fragment loaders (CDNA5 wave32 layouts, CDNA5 ISA §7.12.2)
// A 16x32 f16: M = lane&15; per lane-half two contiguous 8-half runs at
//              K = g*8 and K = 16+g*8.
// B 32x16 f16: N = lane&15; per lane-half two contiguous 8-half runs at
//              K = g*8 and K = 16+g*8 (same as A).
// C/D 16x16 f32: element i -> M = i + 8*g, N = lane&15.
// ---------------------------------------------------------------------------
static __device__ __forceinline__ v16h load_a_frag(const _Float16* base, int rowBase, int ld, int k0, int lane) {
  int g = lane >> 4, m = lane & 15;
  const _Float16* p = base + (size_t)(rowBase + m) * ld + k0;
  FragU u;
  u.h[0] = *(const v8h*)(p + g * 8);
  u.h[1] = *(const v8h*)(p + 16 + g * 8);
  return u.v;
}
static __device__ __forceinline__ v16h load_b_frag(const _Float16* base, int colBase, int ld, int k0, int lane) {
  int g = lane >> 4, n = lane & 15;
  const _Float16* p = base + (size_t)(colBase + n) * ld + k0 + g * 8;
  FragU u;
  u.h[0] = *(const v8h*)(p);
  u.h[1] = *(const v8h*)(p + 16);
  return u.v;
}
// B operand where the source is stored [K][N] (transposed gather)
static __device__ __forceinline__ v16h load_bt_frag(const _Float16* base, int colBase, int ld, int k0, int lane) {
  int g = lane >> 4, n = lane & 15;
  v16h r;
#pragma unroll
  for (int e = 0; e < 16; ++e)
    r[e] = base[(size_t)(k0 + g * 8 + (e & 7) + ((e >> 3) << 4)) * ld + colBase + n];
  return r;
}
// A operand where the source is stored [K][M] (transposed gather)
static __device__ __forceinline__ v16h load_at_frag(const _Float16* base, int rowBase, int ld, int k0, int lane) {
  int g = lane >> 4, m = lane & 15;
  v16h r;
#pragma unroll
  for (int e = 0; e < 16; ++e) {
    int j = e >> 1, p = e & 1;
    int kk = ((j >> 2) << 4) + (g << 3) + ((j & 3) << 1) + p;
    r[e] = base[(size_t)(k0 + kk) * ld + rowBase + m];
  }
  return r;
}
static __device__ __forceinline__ v8f wmma16(v16h a, v16h b, v8f c) {
  return __builtin_amdgcn_wmma_f32_16x16x32_f16(false, a, false, b, (short)0, c, false, false);
}

// ---------------------------------------------------------------------------
// fp32 -> f16 pack (one-time operand conversion; halves GEMM HBM traffic)
// ---------------------------------------------------------------------------
__global__ __launch_bounds__(256) void f32_to_f16_kernel(
    const float* __restrict__ x, _Float16* __restrict__ y, int n)
{
  int i = (blockIdx.x * 256 + threadIdx.x) * 8;
  if (i >= n) return;
  float4 a = *(const float4*)(x + i);
  float4 b = *(const float4*)(x + i + 4);
  v8h h;
  h[0] = (_Float16)a.x; h[1] = (_Float16)a.y; h[2] = (_Float16)a.z; h[3] = (_Float16)a.w;
  h[4] = (_Float16)b.x; h[5] = (_Float16)b.y; h[6] = (_Float16)b.z; h[7] = (_Float16)b.w;
  *(v8h*)(y + i) = h;
}

// ---------------------------------------------------------------------------
// GEMM: Y[M,N] = X[M,K] @ W[N,K]^T (+ bias[n]); f16 operands, fp32 out.
// 128x128 block tile, 8 waves (2x4), each wave 64x32 = 4x2 WMMA tiles.
// Tiles staged via async global->LDS b128 copies, double buffered:
// issue stage k+1, s_wait_asynccnt 4 (stage k done), barrier, 8 WMMAs.
// ---------------------------------------------------------------------------
#define GM 128
#define GN 128
#define GK 32

__global__ __launch_bounds__(256) void gemm_xwt_f16_kernel(
    const _Float16* __restrict__ X, const _Float16* __restrict__ W,
    const float* __restrict__ bias, float* __restrict__ Y,
    int M, int N, int K)
{
  __shared__ __attribute__((aligned(16))) _Float16 sA[2][GM * GK];  // 2 x 8 KB
  __shared__ __attribute__((aligned(16))) _Float16 sB[2][GN * GK];  // 2 x 8 KB
  int tid = threadIdx.x, lane = tid & 31, wave = tid >> 5;
  int m0 = blockIdx.y * GM, n0 = blockIdx.x * GN;
  int wm = (wave >> 2) * 64, wn = (wave & 3) * 32;

  // per-thread 16B-chunk coordinates: 512 chunks per 128x32 tile, 2 per thread
  int c0 = tid, c1 = tid + 256;
  int r0c = c0 >> 2, h0 = (c0 & 3) * 8;
  int r1c = c1 >> 2, h1 = (c1 & 3) * 8;
  int wr0 = min(n0 + r0c, N - 1), wr1 = min(n0 + r1c, N - 1);  // clamp; garbage cols masked at store

  v8f zero = {};
  v8f acc[4][2];
#pragma unroll
  for (int r = 0; r < 4; ++r)
#pragma unroll
    for (int c = 0; c < 2; ++c) acc[r][c] = zero;

  auto stage = [&](int buf, int k0) {
    async_ld_b128(lds_off_(&sA[buf][r0c * GK + h0]), X + (size_t)(m0 + r0c) * K + k0 + h0);
    async_ld_b128(lds_off_(&sA[buf][r1c * GK + h1]), X + (size_t)(m0 + r1c) * K + k0 + h1);
    async_ld_b128(lds_off_(&sB[buf][r0c * GK + h0]), W + (size_t)wr0 * K + k0 + h0);
    async_ld_b128(lds_off_(&sB[buf][r1c * GK + h1]), W + (size_t)wr1 * K + k0 + h1);
  };

  int nk = K / GK;
  stage(0, 0);
  for (int ks = 0; ks < nk; ++ks) {
    int buf = ks & 1;
    if (ks + 1 < nk) {          // uniform branch
      stage(buf ^ 1, (ks + 1) * GK);
      WAIT_ASYNC_4();           // our stage-ks copies done; stage ks+1 in flight
    } else {
      WAIT_ASYNC_0();
    }
    __syncthreads();            // all waves' stage-ks copies visible

    v16h bfr[2];
#pragma unroll
    for (int c = 0; c < 2; ++c) bfr[c] = load_b_frag(&sB[buf][0], wn + c * 16, GK, 0, lane);
#pragma unroll
    for (int r = 0; r < 4; ++r) {
      v16h afr = load_a_frag(&sA[buf][0], wm + r * 16, GK, 0, lane);
#pragma unroll
      for (int c = 0; c < 2; ++c) acc[r][c] = wmma16(afr, bfr[c], acc[r][c]);
    }
    __syncthreads();            // buffer free for stage ks+2
  }

  int g = lane >> 4, nn = lane & 15;
#pragma unroll
  for (int r = 0; r < 4; ++r)
#pragma unroll
    for (int c = 0; c < 2; ++c) {
      int gn = n0 + wn + c * 16 + nn;
      if (gn >= N) continue;
      float bv = bias ? bias[gn] : 0.f;
#pragma unroll
      for (int i = 0; i < 8; ++i) {
        int gm = m0 + wm + r * 16 + g * 8 + i;
        if (gm < M) Y[(size_t)gm * N + gn] = acc[r][c][i] + bv;
      }
    }
}

// ---------------------------------------------------------------------------
// Short causal conv (K=4) + SiLU, channels-last
// ---------------------------------------------------------------------------
__global__ __launch_bounds__(256) void shortconv_silu_kernel(
    const float* __restrict__ x, const float* __restrict__ w,
    float* __restrict__ y, int C)
{
  int idx = blockIdx.x * 256 + threadIdx.x;
  if (idx >= BT * C) return;
  int c = idx % C, bt = idx / C;
  int t = bt % T_, b = bt / T_;
  float acc = 0.f;
#pragma unroll
  for (int j = 0; j < 4; ++j) {
    int ts = t - 3 + j;
    if (ts >= 0) acc += x[((size_t)(b * T_ + ts)) * C + c] * w[c * 4 + j];
  }
  y[idx] = acc * sigmoidf_(acc);
}

// ---------------------------------------------------------------------------
// L2 normalize rows of length 128 (one wave per row)
// ---------------------------------------------------------------------------
__global__ __launch_bounds__(256) void l2norm128_kernel(float* __restrict__ x, int rows)
{
  int row = blockIdx.x * 8 + (threadIdx.x >> 5);
  if (row >= rows) return;
  int lane = threadIdx.x & 31;
  float4* p = (float4*)(x + (size_t)row * 128);
  float4 vv = p[lane];
  float s = vv.x * vv.x + vv.y * vv.y + vv.z * vv.z + vv.w * vv.w;
#pragma unroll
  for (int m = 16; m >= 1; m >>= 1) s += __shfl_xor(s, m, 32);
  float inv = 1.f / fmaxf(sqrtf(s), 1e-12f);
  vv.x *= inv; vv.y *= inv; vv.z *= inv; vv.w *= inv;
  p[lane] = vv;
}

__global__ __launch_bounds__(256) void sigmoid_kernel(float* __restrict__ x, int n)
{
  int i = blockIdx.x * 256 + threadIdx.x;
  if (i < n) x[i] = sigmoidf_(x[i]);
}

// ---------------------------------------------------------------------------
// log_alpha = -exp(A_log[h]) * softplus(graw + dt_bias); per-chunk cumsum -> G
// ---------------------------------------------------------------------------
__global__ __launch_bounds__(128) void gating_kernel(
    float* __restrict__ G, const float* __restrict__ A_log, const float* __restrict__ dtb)
{
  int blk = blockIdx.x;
  int h = blk % H_, nc = (blk / H_) % NCHUNK, b = blk / (H_ * NCHUNK);
  int d = threadIdx.x;
  float negA = -__expf(A_log[h]);
  float bias = dtb[h * DK_ + d];
  float run = 0.f;
  for (int t = 0; t < CHUNK_; ++t) {
    size_t idx = ((size_t)(b * T_ + nc * CHUNK_ + t)) * (H_ * DK_) + h * DK_ + d;
    run += negA * softplusf_(G[idx] + bias);
    G[idx] = run;
  }
}

// ---------------------------------------------------------------------------
// Chunked multi-key delta-attention scan. One block (8 waves) per (b,h);
// S^T (128x128) resident in LDS for all 32 chunks. Decay factorized so
// M / attn / solve-updates / state-update are all f16 WMMA GEMMs.
// ---------------------------------------------------------------------------
__global__ __launch_bounds__(256) void mkda_scan_kernel(
    const float* __restrict__ q, const float* __restrict__ k, const float* __restrict__ v,
    const float* __restrict__ G, const float* __restrict__ beta, float* __restrict__ o)
{
  extern __shared__ char smem_raw[];
  _Float16* kexpH = (_Float16*)smem_raw;               // [256][128]
  _Float16* wdecH = kexpH + CR_ * DK_;                 // [256][128]
  _Float16* uH    = wdecH + CR_ * DK_;                 // [256][128]
  _Float16* qexpH = uH + CR_ * DK_;                    // [64][128]
  _Float16* ShT   = qexpH + CHUNK_ * DK_;              // [128][128], S^T: [v][d]
  char* scratch   = (char*)(ShT + DV_ * DK_);
  _Float16* MrowH = (_Float16*)scratch;                // [32][256]
  float*    rhsF  = (float*)(scratch + 32 * CR_ * sizeof(_Float16)); // [32][128]
  _Float16* attnH = (_Float16*)scratch;                // [64][256] (overlays Mrow+rhs)
  float*    eg    = (float*)(scratch + 64 * CR_ * sizeof(_Float16)); // [128]

  int tid = threadIdx.x, lane = tid & 31, wave = tid >> 5;
  int b = blockIdx.x / H_, h = blockIdx.x % H_;
  v8f zero = {};

  for (int i = tid; i < DV_ * DK_; i += 256) ShT[i] = (_Float16)0.f;
  __syncthreads();

  for (int nc = 0; nc < NCHUNK; ++nc) {
    int t0 = nc * CHUNK_;

    // ---- Phase 1: stage kexp / wdec / qexp / e^{g_last} ----
    for (int idx = tid; idx < CR_ * DK_; idx += 256) {
      int ti = idx >> 7, d = idx & 127;
      int t = ti >> 2, i = ti & 3;
      size_t gi = ((size_t)(b * T_ + t0 + t)) * (H_ * DK_) + h * DK_ + d;
      float gt = G[gi];
      size_t ki = ((size_t)(b * T_ + t0 + t)) * (H_ * R_ * DK_) + (h * R_ + i) * DK_ + d;
      float kv = k[ki];
      size_t bi = ((size_t)(b * T_ + t0 + t)) * (H_ * R_) + h * R_ + i;
      kexpH[idx] = (_Float16)(kv * __expf(gt));
      wdecH[idx] = (_Float16)(beta[bi] * kv * __expf(-gt));
    }
    for (int idx = tid; idx < CHUNK_ * DK_; idx += 256) {
      int t = idx >> 7, d = idx & 127;
      size_t gi = ((size_t)(b * T_ + t0 + t)) * (H_ * DK_) + h * DK_ + d;
      qexpH[idx] = (_Float16)(q[gi] * __expf(G[gi]));
    }
    if (tid < DK_) {
      size_t gi = ((size_t)(b * T_ + t0 + CHUNK_ - 1)) * (H_ * DK_) + h * DK_ + tid;
      eg[tid] = __expf(G[gi]);
    }
    __syncthreads();

    // ---- Phase 2: blocked forward substitution, 8 row-blocks of 32 ----
    for (int rb = 0; rb < 8; ++rb) {
      int r0 = rb * 32;
      int colTiles = 2 * (rb + 1);
      for (int tile = wave; tile < 2 * colTiles; tile += 8) {
        int rt = tile / colTiles, ct = tile % colTiles;
        v8f acc = zero;
#pragma unroll
        for (int ks = 0; ks < 4; ++ks) {
          v16h a  = load_a_frag(kexpH, r0 + rt * 16, DK_, ks * 32, lane);
          v16h bb = load_b_frag(wdecH, ct * 16, DK_, ks * 32, lane);
          acc = wmma16(a, bb, acc);
        }
        int g = lane >> 4, nn = lane & 15, gc = ct * 16 + nn;
#pragma unroll
        for (int i2 = 0; i2 < 8; ++i2) {
          int lr = rt * 16 + g * 8 + i2;
          MrowH[lr * CR_ + gc] = (_Float16)((gc < r0 + lr) ? acc[i2] : 0.f);
        }
      }
      for (int tile = wave; tile < 16; tile += 8) {
        int rt = tile >> 3, ct = tile & 7;
        v8f acc = zero;
#pragma unroll
        for (int ks = 0; ks < 4; ++ks) {
          v16h a  = load_a_frag(kexpH, r0 + rt * 16, DK_, ks * 32, lane);
          v16h bb = load_b_frag(ShT, ct * 16, DK_, ks * 32, lane);
          acc = wmma16(a, bb, acc);
        }
        int g = lane >> 4, nn = lane & 15, vd = ct * 16 + nn;
#pragma unroll
        for (int i2 = 0; i2 < 8; ++i2) {
          int lr = rt * 16 + g * 8 + i2;
          int ti = r0 + lr, t = ti >> 2, ii = ti & 3;
          size_t vi = ((size_t)(b * T_ + t0 + t)) * (H_ * R_ * DV_) + (h * R_ + ii) * DV_ + vd;
          rhsF[lr * DV_ + vd] = v[vi] - acc[i2];
        }
      }
      __syncthreads();
      if (rb > 0) {
        for (int tile = wave; tile < 16; tile += 8) {
          int rt = tile >> 3, ct = tile & 7;
          v8f acc = zero;
          for (int pb = 0; pb < rb; ++pb) {
            v16h a  = load_a_frag(MrowH, rt * 16, CR_, pb * 32, lane);
            v16h bb = load_bt_frag(uH, ct * 16, DV_, pb * 32, lane);
            acc = wmma16(a, bb, acc);
          }
          int g = lane >> 4, nn = lane & 15, vd = ct * 16 + nn;
#pragma unroll
          for (int i2 = 0; i2 < 8; ++i2)
            rhsF[(rt * 16 + g * 8 + i2) * DV_ + vd] -= acc[i2];
        }
      }
      __syncthreads();
      if (tid < DV_) {
        int vd = tid;
#pragma unroll 1
        for (int r = 1; r < 32; ++r) {
          float s = 0.f;
          for (int c = 0; c < r; ++c)
            s += (float)MrowH[r * CR_ + r0 + c] * rhsF[c * DV_ + vd];
          rhsF[r * DV_ + vd] -= s;
        }
      }
      __syncthreads();
      for (int idx = tid; idx < 32 * DV_; idx += 256)
        uH[(r0 + (idx >> 7)) * DV_ + (idx & 127)] = (_Float16)rhsF[idx];
      __syncthreads();
    }

    // ---- Phase 3a: attn = qexp @ wdec^T, masked s<=t ----
    for (int tile = wave; tile < 64; tile += 8) {
      int rt = tile >> 4, ct = tile & 15;
      v8f acc = zero;
#pragma unroll
      for (int ks = 0; ks < 4; ++ks) {
        v16h a  = load_a_frag(qexpH, rt * 16, DK_, ks * 32, lane);
        v16h bb = load_b_frag(wdecH, ct * 16, DK_, ks * 32, lane);
        acc = wmma16(a, bb, acc);
      }
      int g = lane >> 4, nn = lane & 15, sm = ct * 16 + nn;
#pragma unroll
      for (int i2 = 0; i2 < 8; ++i2) {
        int t = rt * 16 + g * 8 + i2;
        attnH[t * CR_ + sm] = (_Float16)(((sm >> 2) <= t) ? acc[i2] : 0.f);
      }
    }
    __syncthreads();
    // ---- Phase 3b: o = SCALE * (qexp @ S + attn @ u) ----
    for (int tile = wave; tile < 32; tile += 8) {
      int rt = tile >> 3, ct = tile & 7;
      v8f acc = zero;
#pragma unroll
      for (int ks = 0; ks < 4; ++ks) {
        v16h a  = load_a_frag(qexpH, rt * 16, DK_, ks * 32, lane);
        v16h bb = load_b_frag(ShT, ct * 16, DK_, ks * 32, lane);
        acc = wmma16(a, bb, acc);
      }
#pragma unroll
      for (int ks = 0; ks < 8; ++ks) {
        v16h a  = load_a_frag(attnH, rt * 16, CR_, ks * 32, lane);
        v16h bb = load_bt_frag(uH, ct * 16, DV_, ks * 32, lane);
        acc = wmma16(a, bb, acc);
      }
      int g = lane >> 4, nn = lane & 15, vd = ct * 16 + nn;
#pragma unroll
      for (int i2 = 0; i2 < 8; ++i2) {
        int t = rt * 16 + g * 8 + i2;
        size_t oi = ((size_t)(b * T_ + t0 + t)) * (H_ * DV_) + h * DV_ + vd;
        o[oi] = SCALE_ * acc[i2];
      }
    }
    __syncthreads();
    // ---- Phase 4: S^T[v][d] = e^{g_last}[d] * (S^T[v][d] + (wdec^T @ u)[d][v]) ----
    for (int tile = wave; tile < 64; tile += 8) {
      int rt = tile >> 3, ct = tile & 7;
      v8f acc = zero;
#pragma unroll
      for (int ks = 0; ks < 8; ++ks) {
        v16h a  = load_at_frag(wdecH, rt * 16, DK_, ks * 32, lane);
        v16h bb = load_bt_frag(uH, ct * 16, DV_, ks * 32, lane);
        acc = wmma16(a, bb, acc);
      }
      int g = lane >> 4, nn = lane & 15, vd = ct * 16 + nn;
#pragma unroll
      for (int i2 = 0; i2 < 8; ++i2) {
        int d = rt * 16 + g * 8 + i2;
        float oldv = (float)ShT[vd * DK_ + d];
        ShT[vd * DK_ + d] = (_Float16)(eg[d] * (oldv + acc[i2]));
      }
    }
    __syncthreads();
  }
}

// ---------------------------------------------------------------------------
// RMS-norm over DV + onorm_w + sigmoid(gate), in place over gate buffer
// ---------------------------------------------------------------------------
__global__ __launch_bounds__(256) void onorm_gate_kernel(
    const float* __restrict__ o, float* __restrict__ gate, const float* __restrict__ w)
{
  int row = blockIdx.x * 8 + (threadIdx.x >> 5);  // rows = BT*H
  int lane = threadIdx.x & 31;
  const float4* po = (const float4*)(o + (size_t)row * DV_);
  float4* pg = (float4*)(gate + (size_t)row * DV_);
  float4 ov = po[lane];
  float s = ov.x * ov.x + ov.y * ov.y + ov.z * ov.z + ov.w * ov.w;
#pragma unroll
  for (int m = 16; m >= 1; m >>= 1) s += __shfl_xor(s, m, 32);
  float rstd = rsqrtf(s * (1.f / DV_) + 1e-5f);
  float4 gv = pg[lane];
  float4 wv = ((const float4*)w)[lane];
  float4 out;
  out.x = ov.x * rstd * wv.x * sigmoidf_(gv.x);
  out.y = ov.y * rstd * wv.y * sigmoidf_(gv.y);
  out.z = ov.z * rstd * wv.z * sigmoidf_(gv.z);
  out.w = ov.w * rstd * wv.w * sigmoidf_(gv.w);
  pg[lane] = out;
}

// ---------------------------------------------------------------------------
// Host orchestration
// ---------------------------------------------------------------------------
extern "C" void kernel_launch(void* const* d_in, const int* in_sizes, int n_in,
                              void* d_out, int out_size, void* d_ws, size_t ws_size,
                              hipStream_t stream)
{
  (void)in_sizes; (void)n_in; (void)out_size; (void)ws_size;
  const float* HS    = (const float*)d_in[0];
  const float* Wq    = (const float*)d_in[1];
  const float* Wk    = (const float*)d_in[2];
  const float* Wv    = (const float*)d_in[3];
  const float* cq    = (const float*)d_in[4];
  const float* ck    = (const float*)d_in[5];
  const float* cv    = (const float*)d_in[6];
  const float* Wf1   = (const float*)d_in[7];
  const float* Wf2   = (const float*)d_in[8];
  const float* Wb    = (const float*)d_in[9];
  const float* A_log = (const float*)d_in[10];
  const float* dtb   = (const float*)d_in[11];
  const float* Wg1   = (const float*)d_in[12];
  const float* bg1   = (const float*)d_in[13];
  const float* Wg2   = (const float*)d_in[14];
  const float* bg2   = (const float*)d_in[15];
  const float* onw   = (const float*)d_in[16];
  const float* Wo    = (const float*)d_in[17];
  float* out = (float*)d_out;

  size_t off = 0;
  auto alloc = [&](size_t bytes) -> char* {
    char* p = (char*)d_ws + off;
    off += (bytes + 255) & ~(size_t)255;
    return p;
  };
  // fp32 working buffers
  float* pre   = (float*)alloc((size_t)BT * 8192 * 4);
  float* kb    = (float*)alloc((size_t)BT * 8192 * 4);
  float* vb    = (float*)alloc((size_t)BT * 8192 * 4);
  float* qb    = (float*)alloc((size_t)BT * 2048 * 4);
  float* Gb    = (float*)alloc((size_t)BT * 2048 * 4);
  float* betab = (float*)alloc((size_t)BT * 64 * 4);
  float* f1b   = (float*)alloc((size_t)BT * 128 * 4);
  float* ob    = (float*)alloc((size_t)BT * 2048 * 4);
  float* gateb = (float*)alloc((size_t)BT * 2048 * 4);
  // f16 GEMM operands (one-time conversion)
  _Float16* HSh  = (_Float16*)alloc((size_t)BT * 2048 * 2);
  _Float16* Wqh  = (_Float16*)alloc((size_t)2048 * 2048 * 2);
  _Float16* Wkh  = (_Float16*)alloc((size_t)8192 * 2048 * 2);
  _Float16* Wvh  = (_Float16*)alloc((size_t)8192 * 2048 * 2);
  _Float16* Wf1h = (_Float16*)alloc((size_t)128 * 2048 * 2);
  _Float16* Wf2h = (_Float16*)alloc((size_t)2048 * 128 * 2);
  _Float16* Wbh  = (_Float16*)alloc((size_t)64 * 2048 * 2);
  _Float16* Wg1h = (_Float16*)alloc((size_t)128 * 2048 * 2);
  _Float16* Wg2h = (_Float16*)alloc((size_t)2048 * 128 * 2);
  _Float16* Woh  = (_Float16*)alloc((size_t)2048 * 2048 * 2);
  _Float16* f1h  = (_Float16*)alloc((size_t)BT * 128 * 2);
  _Float16* gth  = (_Float16*)alloc((size_t)BT * 2048 * 2);

  dim3 blk(256);
  auto cvt = [&](const float* x, _Float16* y, size_t n) {
    f32_to_f16_kernel<<<dim3((unsigned)((n / 8 + 255) / 256)), blk, 0, stream>>>(x, y, (int)n);
  };
  auto gemm = [&](const _Float16* X, const _Float16* W, const float* bias, float* Y,
                  int M, int N, int K) {
    dim3 grid((N + GN - 1) / GN, (M + GM - 1) / GM);
    gemm_xwt_f16_kernel<<<grid, blk, 0, stream>>>(X, W, bias, Y, M, N, K);
  };

  // one-time operand conversions
  cvt(HS, HSh, (size_t)BT * 2048);
  cvt(Wq, Wqh, (size_t)2048 * 2048);
  cvt(Wk, Wkh, (size_t)8192 * 2048);
  cvt(Wv, Wvh, (size_t)8192 * 2048);
  cvt(Wf1, Wf1h, (size_t)128 * 2048);
  cvt(Wf2, Wf2h, (size_t)2048 * 128);
  cvt(Wb, Wbh, (size_t)64 * 2048);
  cvt(Wg1, Wg1h, (size_t)128 * 2048);
  cvt(Wg2, Wg2h, (size_t)2048 * 128);
  cvt(Wo, Woh, (size_t)2048 * 2048);

  // k path
  gemm(HSh, Wkh, nullptr, pre, BT, 8192, 2048);
  shortconv_silu_kernel<<<dim3((BT * 8192 + 255) / 256), blk, 0, stream>>>(pre, ck, kb, 8192);
  l2norm128_kernel<<<dim3(BT * H_ * R_ / 8), blk, 0, stream>>>(kb, BT * H_ * R_);
  // v path
  gemm(HSh, Wvh, nullptr, pre, BT, 8192, 2048);
  shortconv_silu_kernel<<<dim3((BT * 8192 + 255) / 256), blk, 0, stream>>>(pre, cv, vb, 8192);
  // q path
  gemm(HSh, Wqh, nullptr, pre, BT, 2048, 2048);
  shortconv_silu_kernel<<<dim3((BT * 2048 + 255) / 256), blk, 0, stream>>>(pre, cq, qb, 2048);
  l2norm128_kernel<<<dim3(BT * H_ / 8), blk, 0, stream>>>(qb, BT * H_);
  // decay path
  gemm(HSh, Wf1h, nullptr, f1b, BT, 128, 2048);
  cvt(f1b, f1h, (size_t)BT * 128);
  gemm(f1h, Wf2h, nullptr, Gb, BT, 2048, 128);
  gating_kernel<<<dim3(B_ * NCHUNK * H_), dim3(128), 0, stream>>>(Gb, A_log, dtb);
  // beta
  gemm(HSh, Wbh, nullptr, betab, BT, 64, 2048);
  sigmoid_kernel<<<dim3((BT * 64 + 255) / 256), blk, 0, stream>>>(betab, BT * 64);
  // chunked delta-attention scan
  size_t shmem = (size_t)(3 * CR_ * DK_ + CHUNK_ * DK_ + DV_ * DK_) * sizeof(_Float16)
               + (size_t)64 * CR_ * sizeof(_Float16) + 128 * sizeof(float);  // 279,040 B
  hipFuncSetAttribute((const void*)mkda_scan_kernel,
                      hipFuncAttributeMaxDynamicSharedMemorySize, (int)shmem);
  mkda_scan_kernel<<<dim3(B_ * H_), blk, shmem, stream>>>(qb, kb, vb, Gb, betab, ob);
  // gate path
  gemm(HSh, Wg1h, bg1, f1b, BT, 128, 2048);
  cvt(f1b, f1h, (size_t)BT * 128);
  gemm(f1h, Wg2h, bg2, gateb, BT, 2048, 128);
  onorm_gate_kernel<<<dim3(BT * H_ / 8), blk, 0, stream>>>(ob, gateb, onw);
  cvt(gateb, gth, (size_t)BT * 2048);
  // output projection
  gemm(gth, Woh, nullptr, out, BT, 2048, 2048);
}
